// MultiHeadAttention_14018773254203
// MI455X (gfx1250) — matrix operand order repaired and checked by reading, not compiled, not run on hardware
//
#include <hip/hip_runtime.h>

// ---------------------------------------------------------------------------
// MHA (conv1x1 projections + softmax-over-query-axis attention) for gfx1250.
// All GEMMs via v_wmma_f32_16x16x32_bf16 (wave32).
//
// Pipeline (all on `stream`):
//   cvtT(q) -> qsT bf16 [b][l][c]      (B-operand layout for projections)
//   cvtT(k) -> ksT bf16 [b][l][c]
//   cvtT(v) -> vsT bf16 [b][l][c]
//   cvt(Wq/Wk/Wv) -> bf16 [o][c]       (A-operand layout)
//   proj -> qT [b][l][c], kT [b][l][c], vN [b][c][l]   (bf16)
//   stats: per (bh,s): m_s, 1/Z_s over the query axis t (online softmax)
//   out:   out[c,t] = sum_s exp(score*SCALE - m_s)/Z_s * v[c,s]  via WMMA
// ---------------------------------------------------------------------------

constexpr int   B_ = 8;
constexpr int   C_ = 1024;
constexpr int   L_ = 1024;
constexpr int   H_ = 16;
constexpr float SCALE_ = 0.125f;   // 1/sqrt(64)

typedef __attribute__((ext_vector_type(16))) __bf16        v16bf;
typedef __attribute__((ext_vector_type(8)))  float         v8f;
typedef __attribute__((ext_vector_type(4)))  unsigned int  u32x4;
typedef __attribute__((ext_vector_type(4)))  float         f32x4;

union FragAB {
  unsigned int u[8];
  u32x4        q[2];
  v16bf        v;
};
union FragC {
  float f[8];
  v8f   v;
};

// Native conversion: single v_cvt instead of manual bfe/add3/shift RNE.
__device__ __forceinline__ unsigned short f2bf(float x) {
  return __builtin_bit_cast(unsigned short, (__bf16)x);
}
__device__ __forceinline__ unsigned int pack2bf(float a, float b) {
  return ((unsigned int)f2bf(b) << 16) | f2bf(a);
}

__device__ __forceinline__ v8f wmma_bf16(v16bf a, v16bf b, v8f c) {
  // (neg_a, A, neg_b, B, c_mod, C, reuse_a, reuse_b)
  return __builtin_amdgcn_wmma_f32_16x16x32_bf16(false, a, false, b,
                                                 (short)0, c, false, false);
}

// B-fragment (32x16, K-major pairs): 16 contiguous bf16 (32B, 16B aligned).
__device__ __forceinline__ void load16(FragAB& f, const unsigned short* p) {
  const u32x4* q = (const u32x4*)p;
  f.q[0] = q[0];
  f.q[1] = q[1];
}

// A-fragment (16x32): per-lane row, 16 contiguous bf16 at 2*klo (k = 16h + s).
__device__ __forceinline__ void loadA(FragAB& f, const unsigned short* row, int klo) {
  f.q[0] = *(const u32x4*)(row + 2 * klo);
  f.q[1] = *(const u32x4*)(row + 2 * klo + 8);
}

// ---------------------------------------------------------------------------
// Pre-pass: fp32 [b][c][l] -> bf16 [b][l][c]  (64x64 tile transpose via LDS).
// ---------------------------------------------------------------------------
__global__ __launch_bounds__(256)
void cvt_transpose_kernel(const float* __restrict__ X,
                          unsigned short* __restrict__ XT) {
  __shared__ unsigned short t[64][72];   // pitch 72 shorts = 144B (16B multiple)
  const int lb = blockIdx.x * 64;
  const int cb = blockIdx.y * 64;
  const int b  = blockIdx.z;
  const int tid = threadIdx.x;

  const int cr = tid >> 2;            // 0..63 source channel row
  const int lc = (tid & 3) * 16;      // 0,16,32,48
  const float* src = X + ((size_t)b * C_ + cb + cr) * L_ + lb + lc;
#pragma unroll
  for (int j = 0; j < 16; ++j) t[lc + j][cr] = f2bf(src[j]);
  __syncthreads();

  const int lr = tid >> 2;            // 0..63 dest l row
  const int cc = (tid & 3) * 16;
  unsigned short* dst = XT + ((size_t)b * L_ + lb + lr) * C_ + cb + cc;
  *(u32x4*)(dst)     = *(const u32x4*)&t[lr][cc];
  *(u32x4*)(dst + 8) = *(const u32x4*)&t[lr][cc + 8];
}

// ---------------------------------------------------------------------------
// Pre-pass: elementwise fp32 -> bf16 (weights), 8 elems/thread, vectorized.
// ---------------------------------------------------------------------------
__global__ __launch_bounds__(256)
void cvt_kernel(const float* __restrict__ X, unsigned short* __restrict__ Y,
                int n) {
  const int i = (blockIdx.x * 256 + threadIdx.x) * 8;
  if (i >= n) return;
  const f32x4 x0 = *(const f32x4*)(X + i);
  const f32x4 x1 = *(const f32x4*)(X + i + 4);
  union { unsigned int u[4]; u32x4 q; } o;
  o.u[0] = pack2bf(x0.x, x0.y);
  o.u[1] = pack2bf(x0.z, x0.w);
  o.u[2] = pack2bf(x1.x, x1.y);
  o.u[3] = pack2bf(x1.z, x1.w);
  *(u32x4*)(Y + i) = o.q;
}

// ---------------------------------------------------------------------------
// Projection: Y[b,o,l] = sum_c W[o,c] * X[b,c,l] + bias[o], bf16 in/out.
// Block tile 128(o) x 128(l); wave tile 32(o) x 64(l): 8 WMMA / 6 frag loads
// per K-step. No LDS, no barriers: pure b128 fragment loads + WMMA.
// ---------------------------------------------------------------------------
__global__ __launch_bounds__(256)
void proj_kernel(const unsigned short* __restrict__ XT,  // [b][l][c] bf16
                 const unsigned short* __restrict__ Wb,  // [o][c]    bf16
                 const float* __restrict__ bias,
                 unsigned short* __restrict__ Y,
                 int transpose_out) {
  const int lb   = blockIdx.x * 128;
  const int ob   = blockIdx.y * 128;
  const int b    = blockIdx.z;
  const int tid  = threadIdx.x;
  const int lane = tid & 31;
  const int wave = tid >> 5;
  const int l15  = lane & 15;
  const int lh   = lane >> 4;
  const int mo   = (wave & 3) * 32;   // o offset of this wave
  const int nl   = (wave >> 2) * 64;  // l offset of this wave

  FragC acc[2][4];
#pragma unroll
  for (int mi = 0; mi < 2; ++mi)
#pragma unroll
    for (int ni = 0; ni < 4; ++ni)
#pragma unroll
      for (int i = 0; i < 8; ++i) acc[mi][ni].f[i] = 0.f;

  const unsigned short* wrow0 = Wb + (size_t)(ob + mo + l15) * C_;
  const unsigned short* wrow1 = Wb + (size_t)(ob + mo + 16 + l15) * C_;
  const unsigned short* xrow[4];
#pragma unroll
  for (int ni = 0; ni < 4; ++ni)
    xrow[ni] = XT + ((size_t)b * L_ + lb + nl + ni * 16 + l15) * C_;

  const int klo = lh * 8;
  const int kbh = lh * 16;

  for (int kt = 0; kt < C_; kt += 32) {
    __builtin_prefetch(wrow0 + kt + 128, 0, 1);
    __builtin_prefetch(xrow[0] + kt + 128, 0, 1);

    FragAB a0, a1;
    loadA(a0, wrow0 + kt, klo);
    loadA(a1, wrow1 + kt, klo);
    FragAB bfr[4];
#pragma unroll
    for (int ni = 0; ni < 4; ++ni) load16(bfr[ni], xrow[ni] + kt + kbh);

#pragma unroll
    for (int ni = 0; ni < 4; ++ni) {
      acc[0][ni].v = wmma_bf16(a0.v, bfr[ni].v, acc[0][ni].v);
      acc[1][ni].v = wmma_bf16(a1.v, bfr[ni].v, acc[1][ni].v);
    }
  }

  // epilogue: add bias, convert, store
  float barr[2][8];
#pragma unroll
  for (int mi = 0; mi < 2; ++mi)
#pragma unroll
    for (int r = 0; r < 8; ++r)
      barr[mi][r] = bias[ob + mo + mi * 16 + 8 * lh + r];

#pragma unroll
  for (int mi = 0; mi < 2; ++mi) {
    const int obase = ob + mo + mi * 16 + 8 * lh;
#pragma unroll
    for (int ni = 0; ni < 4; ++ni) {
      const int l = lb + nl + ni * 16 + l15;
      if (transpose_out) {
        // consecutive r -> consecutive channel o: one 16B packed store
        union { unsigned int u[4]; u32x4 q; } pk;
#pragma unroll
        for (int rr = 0; rr < 4; ++rr)
          pk.u[rr] = pack2bf(acc[mi][ni].f[2 * rr] + barr[mi][2 * rr],
                             acc[mi][ni].f[2 * rr + 1] + barr[mi][2 * rr + 1]);
        *(u32x4*)(Y + ((size_t)b * L_ + l) * C_ + obase) = pk.q;
      } else {
#pragma unroll
        for (int r = 0; r < 8; ++r)
          Y[((size_t)b * C_ + obase + r) * L_ + l] =
              f2bf(acc[mi][ni].f[r] + barr[mi][r]);
      }
    }
  }
}

// ---------------------------------------------------------------------------
// Stats: per (bh, s) column, online softmax stats over all t (query axis).
// Block = 64 s-columns; 8 waves stripe over t in chunks of 16.
// ---------------------------------------------------------------------------
__global__ __launch_bounds__(256)
void attn_stats_kernel(const unsigned short* __restrict__ qT,
                       const unsigned short* __restrict__ kT,
                       float* __restrict__ Mstat, float* __restrict__ Istat) {
  const int sb   = blockIdx.x * 64;
  const int bh   = blockIdx.y;
  const int b    = bh >> 4;
  const int hc   = (bh & 15) * 64;
  const int tid  = threadIdx.x;
  const int lane = tid & 31;
  const int wave = tid >> 5;
  const int l15  = lane & 15;
  const int lh   = lane >> 4;

  __shared__ float mS[8][64];
  __shared__ float zS[8][64];

  // K fragments invariant over the t loop: keep in registers.
  FragAB kf[4][2];
#pragma unroll
  for (int n = 0; n < 4; ++n) {
    const unsigned short* krow =
        kT + ((size_t)b * L_ + sb + n * 16 + l15) * C_ + hc;
    load16(kf[n][0], krow + lh * 16);
    load16(kf[n][1], krow + 32 + lh * 16);
  }

  float m_run[4], z_run[4];
#pragma unroll
  for (int n = 0; n < 4; ++n) { m_run[n] = -1e30f; z_run[n] = 0.f; }

  for (int tb = wave * 16; tb < L_; tb += 128) {
    FragAB a0, a1;
    const unsigned short* qrow = qT + ((size_t)b * L_ + tb + l15) * C_ + hc;
    loadA(a0, qrow, lh * 8);
    loadA(a1, qrow + 32, lh * 8);
#pragma unroll
    for (int n = 0; n < 4; ++n) {
      FragC c;
#pragma unroll
      for (int i = 0; i < 8; ++i) c.f[i] = 0.f;
      c.v = wmma_bf16(a0.v, kf[n][0].v, c.v);
      c.v = wmma_bf16(a1.v, kf[n][1].v, c.v);

      float tmax = -1e30f;
#pragma unroll
      for (int r = 0; r < 8; ++r) tmax = fmaxf(tmax, c.f[r] * SCALE_);
      tmax = fmaxf(tmax, __shfl_xor(tmax, 16, 32));
      const float m_new = fmaxf(m_run[n], tmax);
      float z = 0.f;
#pragma unroll
      for (int r = 0; r < 8; ++r) z += __expf(c.f[r] * SCALE_ - m_new);
      z += __shfl_xor(z, 16, 32);
      z_run[n] = z_run[n] * __expf(m_run[n] - m_new) + z;
      m_run[n] = m_new;
    }
  }

  if (lane < 16) {
#pragma unroll
    for (int n = 0; n < 4; ++n) {
      mS[wave][n * 16 + l15] = m_run[n];
      zS[wave][n * 16 + l15] = z_run[n];
    }
  }
  __syncthreads();
  if (tid < 64) {
    float m = mS[0][tid], z = zS[0][tid];
#pragma unroll
    for (int w = 1; w < 8; ++w) {
      const float mw2 = mS[w][tid];
      const float mn  = fmaxf(m, mw2);
      z = z * __expf(m - mn) + zS[w][tid] * __expf(mw2 - mn);
      m = mn;
    }
    Mstat[(size_t)bh * L_ + sb + tid] = m;
    Istat[(size_t)bh * L_ + sb + tid] = 1.f / z;
  }
}

// ---------------------------------------------------------------------------
// Output: out[c,t] = sum_s exp(score*SCALE - m_s) * inv_s * v[c,s].
// Each wave: 16 t-columns, full 64 c-rows, K-loop over s in chunks of 32.
// Score tile (WMMA, C-layout) -> exp -> bf16 -> per-wave LDS -> B-fragment.
// ---------------------------------------------------------------------------
__global__ __launch_bounds__(256)
void attn_out_kernel(const unsigned short* __restrict__ qT,
                     const unsigned short* __restrict__ kT,
                     const unsigned short* __restrict__ Vb,
                     const float* __restrict__ Mstat,
                     const float* __restrict__ Istat,
                     float* __restrict__ Out) {
  const int bh   = blockIdx.y;
  const int b    = bh >> 4;
  const int hc   = (bh & 15) * 64;
  const int tid  = threadIdx.x;
  const int lane = tid & 31;
  const int wave = tid >> 5;
  const int l15  = lane & 15;
  const int lh   = lane >> 4;
  const int tb   = blockIdx.x * 128 + wave * 16;

  __shared__ unsigned short pl[8][16][32];   // per-wave 16t x 32s prob tile

  FragAB a0, a1;   // Q fragments: fixed t rows for whole kernel
  const unsigned short* qrow = qT + ((size_t)b * L_ + tb + l15) * C_ + hc;
  loadA(a0, qrow, lh * 8);
  loadA(a1, qrow + 32, lh * 8);

  FragC acc[4];
#pragma unroll
  for (int m = 0; m < 4; ++m)
#pragma unroll
    for (int i = 0; i < 8; ++i) acc[m].f[i] = 0.f;

  const float* Mrow = Mstat + (size_t)bh * L_;
  const float* Irow = Istat + (size_t)bh * L_;

  for (int sc = 0; sc < L_; sc += 32) {
    __builtin_prefetch(kT + ((size_t)b * L_ + sc + 32 + l15) * C_ + hc, 0, 1);

    FragAB k00, k01, k10, k11;
    const unsigned short* kr0 = kT + ((size_t)b * L_ + sc + l15) * C_ + hc;
    const unsigned short* kr1 = kT + ((size_t)b * L_ + sc + 16 + l15) * C_ + hc;
    load16(k00, kr0 + lh * 16);
    load16(k01, kr0 + 32 + lh * 16);
    load16(k10, kr1 + lh * 16);
    load16(k11, kr1 + 32 + lh * 16);

    FragC s0, s1;
#pragma unroll
    for (int i = 0; i < 8; ++i) { s0.f[i] = 0.f; s1.f[i] = 0.f; }
    s0.v = wmma_bf16(a0.v, k00.v, s0.v);
    s0.v = wmma_bf16(a1.v, k01.v, s0.v);
    s1.v = wmma_bf16(a0.v, k10.v, s1.v);
    s1.v = wmma_bf16(a1.v, k11.v, s1.v);

    const float m0 = Mrow[sc + l15],      i0 = Irow[sc + l15];
    const float m1 = Mrow[sc + 16 + l15], i1 = Irow[sc + 16 + l15];
#pragma unroll
    for (int r = 0; r < 8; ++r) {
      const int t_local = r + 8 * lh;     // C-layout row for this lane half
      pl[wave][t_local][l15]      = f2bf(__expf(s0.f[r] * SCALE_ - m0) * i0);
      pl[wave][t_local][16 + l15] = f2bf(__expf(s1.f[r] * SCALE_ - m1) * i1);
    }
    // same-wave LDS RAW; compiler inserts the s_wait_dscnt
    FragAB pf;   // B-fragment of P^T: K=s(32), N=t(16)
    load16(pf, &pl[wave][l15][lh * 16]);

#pragma unroll
    for (int m = 0; m < 4; ++m) {
      FragAB vf;
      const unsigned short* vrow =
          Vb + ((size_t)b * C_ + hc + m * 16 + l15) * L_ + sc;
      loadA(vf, vrow, lh * 8);
      acc[m].v = wmma_bf16(vf.v, pf.v, acc[m].v);
    }
  }

#pragma unroll
  for (int m = 0; m < 4; ++m)
#pragma unroll
    for (int r = 0; r < 8; ++r) {
      const int c = hc + m * 16 + r + 8 * lh;
      Out[((size_t)b * C_ + c) * L_ + tb + l15] = acc[m].f[r];
    }
}

// ---------------------------------------------------------------------------
extern "C" void kernel_launch(void* const* d_in, const int* in_sizes, int n_in,
                              void* d_out, int out_size, void* d_ws, size_t ws_size,
                              hipStream_t stream) {
  const float* q  = (const float*)d_in[0];
  const float* k  = (const float*)d_in[1];
  const float* v  = (const float*)d_in[2];
  const float* Wq = (const float*)d_in[3];
  const float* bq = (const float*)d_in[4];
  const float* Wk = (const float*)d_in[5];
  const float* bk = (const float*)d_in[6];
  const float* Wv = (const float*)d_in[7];
  const float* bv = (const float*)d_in[8];
  float* out = (float*)d_out;

  // Workspace layout (~103 MB):
  //   [  0M,16M) qsT bf16 [b][l][c]   (converted+transposed input q)
  //   [ 16M,32M) ksT bf16 [b][l][c]
  //   [ 32M,48M) vsT bf16 [b][l][c]
  //   [ 48M,50M) Wq  bf16 [o][c]
  //   [ 50M,52M) Wk  bf16
  //   [ 52M,54M) Wv  bf16
  //   [ 54M,70M) qT  bf16 [b][l][c]   (projection outputs)
  //   [ 70M,86M) kT  bf16 [b][l][c]
  //   [ 86M,102M) vN bf16 [b][c][l]
  //   [102M, +512K) m_s fp32 ; [+512K, +1M) 1/Z_s fp32
  char* ws = (char*)d_ws;
  unsigned short* qsT = (unsigned short*)(ws);
  unsigned short* ksT = (unsigned short*)(ws + ((size_t)16 << 20));
  unsigned short* vsT = (unsigned short*)(ws + ((size_t)32 << 20));
  unsigned short* Wqb = (unsigned short*)(ws + ((size_t)48 << 20));
  unsigned short* Wkb = (unsigned short*)(ws + ((size_t)50 << 20));
  unsigned short* Wvb = (unsigned short*)(ws + ((size_t)52 << 20));
  unsigned short* qT  = (unsigned short*)(ws + ((size_t)54 << 20));
  unsigned short* kT  = (unsigned short*)(ws + ((size_t)70 << 20));
  unsigned short* vN  = (unsigned short*)(ws + ((size_t)86 << 20));
  float* mstat = (float*)(ws + ((size_t)102 << 20));
  float* istat = (float*)(ws + ((size_t)102 << 20) + ((size_t)512 << 10));

  const dim3 blk(256);
  const dim3 tgrid(L_ / 64, C_ / 64, B_);
  cvt_transpose_kernel<<<tgrid, blk, 0, stream>>>(q, qsT);
  cvt_transpose_kernel<<<tgrid, blk, 0, stream>>>(k, ksT);
  cvt_transpose_kernel<<<tgrid, blk, 0, stream>>>(v, vsT);
  cvt_kernel<<<dim3((C_ * C_) / 2048), blk, 0, stream>>>(Wq, Wqb, C_ * C_);
  cvt_kernel<<<dim3((C_ * C_) / 2048), blk, 0, stream>>>(Wk, Wkb, C_ * C_);
  cvt_kernel<<<dim3((C_ * C_) / 2048), blk, 0, stream>>>(Wv, Wvb, C_ * C_);

  const dim3 pgrid(L_ / 128, C_ / 128, B_);
  proj_kernel<<<pgrid, blk, 0, stream>>>(qsT, Wqb, bq, qT, 1);
  proj_kernel<<<pgrid, blk, 0, stream>>>(ksT, Wkb, bk, kT, 1);
  proj_kernel<<<pgrid, blk, 0, stream>>>(vsT, Wvb, bv, vN, 0);

  attn_stats_kernel<<<dim3(L_ / 64, B_ * H_), blk, 0, stream>>>(qT, kT, mstat, istat);
  attn_out_kernel<<<dim3(L_ / 128, B_ * H_), blk, 0, stream>>>(qT, kT, vN, mstat, istat, out);

  (void)in_sizes; (void)n_in; (void)out_size; (void)ws_size;
}
